// Old_Better_Transformer_49667001811251
// MI455X (gfx1250) — compile-verified, matrix operand order checked by reading, not run on hardware
//
#include <hip/hip_runtime.h>

// ---------------------------------------------------------------------------
// Old_Better_Transformer fused pipeline for MI455X (gfx1250, wave32, WMMA).
//
// Roofline: GEMM = 68.7 GFLOP; fused HBM traffic ~1 GB -> ~44us @ 23.3 TB/s.
// fp32 WMMA (K=4) would be compute bound (~115us); split-bf16 (hi/lo, 3 WMMAs
// per product, f32 accumulate) gives ~fp32 accuracy at ~40us -> memory bound.
// ASYNCcnt-tracked GLOBAL_LOAD_ASYNC_TO_LDS for B tiles + double-buffered LDS.
// ---------------------------------------------------------------------------

#define B_ROWS 16384
#define IN_F   4096
#define P_PART 8
#define D_DIM  512
#define EPS_BN 1e-5f

typedef __bf16 bf16_t;
typedef __attribute__((ext_vector_type(16))) __bf16 v16bf;
typedef __attribute__((ext_vector_type(8)))  __bf16 v8bf;
typedef __attribute__((ext_vector_type(4)))  __bf16 v4bf;
typedef __attribute__((ext_vector_type(8)))  float  v8f;

#ifndef __has_builtin
#define __has_builtin(x) 0
#endif
#if __has_builtin(__builtin_amdgcn_global_load_async_to_lds_b128) && \
    defined(__HIP_DEVICE_COMPILE__)
#define USE_ASYNC_LDS 1
#else
#define USE_ASYNC_LDS 0
#endif

#if USE_ASYNC_LDS
// builtin signature (from clang diag): (int4 AS1* src, int4 AS3* dst, Ii, Ii)
typedef __attribute__((__vector_size__(16))) int av4i;
#define GPTR(x) ((__attribute__((address_space(1))) av4i*)(x))
#define LPTR(x) ((__attribute__((address_space(3))) av4i*)(x))
#endif

// ---------------------------------------------------------------------------
// 1) BatchNorm partial stats: each block owns 256 columns x 256 rows.
//    Deterministic (no float atomics): partials reduced in fixed order later.
// ---------------------------------------------------------------------------
__global__ __launch_bounds__(256)
void bn_partial_kernel(const float* __restrict__ x,
                       float* __restrict__ psum, float* __restrict__ psumsq)
{
    const int col   = blockIdx.x * 256 + threadIdx.x;   // [0, IN)
    const int chunk = blockIdx.y;                       // [0, B/256)
    size_t base = (size_t)chunk * 256 * IN_F + col;
    float s = 0.f, q = 0.f;
    #pragma unroll 4
    for (int r = 0; r < 256; ++r) {
        float v = x[base + (size_t)r * IN_F];
        s += v;
        q += v * v;
    }
    psum  [(size_t)chunk * IN_F + col] = s;
    psumsq[(size_t)chunk * IN_F + col] = q;
}

// ---------------------------------------------------------------------------
// 2) Finalize: ordered reduction of partials -> per-column scale/shift.
// ---------------------------------------------------------------------------
__global__ __launch_bounds__(256)
void bn_finalize_kernel(const float* __restrict__ psum,
                        const float* __restrict__ psumsq,
                        const float* __restrict__ bn_w,
                        const float* __restrict__ bn_b,
                        float* __restrict__ scale,
                        float* __restrict__ shift)
{
    const int c = blockIdx.x * 256 + threadIdx.x;
    float s = 0.f, q = 0.f;
    for (int i = 0; i < B_ROWS / 256; ++i) {   // fixed order -> deterministic
        s += psum  [(size_t)i * IN_F + c];
        q += psumsq[(size_t)i * IN_F + c];
    }
    const float mean = s * (1.0f / B_ROWS);
    const float var  = q * (1.0f / B_ROWS) - mean * mean;
    const float sc   = bn_w[c] * rsqrtf(var + EPS_BN);
    scale[c] = sc;
    shift[c] = bn_b[c] - mean * sc;
}

// ---------------------------------------------------------------------------
// 3) Split weights into bf16 hi/lo, transposed to [p][e][d] (n-major) so the
//    GEMM kernel's global loads and LDS B-tile stores are fully contiguous.
// ---------------------------------------------------------------------------
__global__ __launch_bounds__(256)
void split_weights_kernel(const float* __restrict__ w,
                          bf16_t* __restrict__ whiT, bf16_t* __restrict__ wloT)
{
    const size_t i = (size_t)blockIdx.x * 256 + threadIdx.x;  // (p,e,d) flat
    const int d = (int)(i & (D_DIM - 1));
    const int e = (int)((i >> 9) & (D_DIM - 1));
    const int p = (int)(i >> 18);
    const float v = w[((size_t)p * D_DIM + d) * D_DIM + e];
    const bf16_t hi = (bf16_t)v;
    const bf16_t lo = (bf16_t)(v - (float)hi);
    whiT[i] = hi;
    wloT[i] = lo;
}

// ---------------------------------------------------------------------------
// 4) Fused: normalize -> split-bf16 block GEMM (WMMA) -> bias/gate/residual.
//    Tile: 128(M) x 128(N), K-step 32, double-buffered LDS. 8 waves as
//    2(M) x 4(N); each wave owns 64x32 = 4x2 fragments; 3 WMMAs per frag/K.
// ---------------------------------------------------------------------------
__global__ __launch_bounds__(256)
void fused_bn_blockgemm_gate_kernel(const float*  __restrict__ x,
                                    const bf16_t* __restrict__ whiT,
                                    const bf16_t* __restrict__ wloT,
                                    const float*  __restrict__ scale,
                                    const float*  __restrict__ shift,
                                    const float*  __restrict__ bias,
                                    const float*  __restrict__ gmm,
                                    const float*  __restrict__ bta,
                                    float*        __restrict__ out)
{
    constexpr int BM = 128, BN = 128, BK = 32, LDT = BK + 8; // +8 bf16 pad
    constexpr int NK = D_DIM / BK;                           // 16 K-steps
    __shared__ bf16_t lAhi[2][BM * LDT];
    __shared__ bf16_t lAlo[2][BM * LDT];
    __shared__ bf16_t lBhi[2][BN * LDT];
    __shared__ bf16_t lBlo[2][BN * LDT];

    const int tid  = threadIdx.x;
    const int lane = tid & 31;
    const int wave = tid >> 5;
    const int wm   = (wave >> 2) * 64;   // wave M origin inside tile
    const int wn   = (wave & 3)  * 32;   // wave N origin inside tile
    const int l16  = lane & 15;
    const int kq   = (lane >> 4) * 8;    // WMMA bf16 A/B lane K phase

    const int mBase = blockIdx.x * BM;
    const int nTile = blockIdx.y;        // 32 tiles = 8 partitions * 4
    const int p     = nTile >> 2;
    const int nInP  = (nTile & 3) * BN;  // N origin inside partition

    // cooperative load mappings
    const int aCol  = (tid & 7) * 4;     // float4 within 32-wide K slab
    const int aRow0 = tid >> 3;          // rows aRow0 + 32*i
    const int bD    = (tid & 3) * 8;     // v8bf within 32-wide K slab
    const int bE0   = tid >> 2;          // e rows bE0 + 64*i

    const bf16_t* wHiP = whiT + ((size_t)p * D_DIM + nInP) * D_DIM;
    const bf16_t* wLoP = wloT + ((size_t)p * D_DIM + nInP) * D_DIM;

    const v8f vzero = {0.f, 0.f, 0.f, 0.f, 0.f, 0.f, 0.f, 0.f};
    v8f acc[4][2];
    #pragma unroll
    for (int mf = 0; mf < 4; ++mf)
        #pragma unroll
        for (int nf = 0; nf < 2; ++nf)
            acc[mf][nf] = vzero;

    float4 aReg[4];

    // ---- pipeline stages --------------------------------------------------
    auto loadA = [&](int k0) {                 // global x -> registers
        const int aColG = p * D_DIM + k0 + aCol;
        #pragma unroll
        for (int i = 0; i < 4; ++i)
            aReg[i] = *(const float4*)&x[(size_t)(mBase + aRow0 + 32 * i) * IN_F + aColG];
    };

    auto stageB = [&](int buf, int k0) {       // weights -> LDS (async path)
        #pragma unroll
        for (int i = 0; i < 2; ++i) {
            const int e = bE0 + 64 * i;
            const size_t g = (size_t)e * D_DIM + k0 + bD;
            const int lo = e * LDT + bD;
#if USE_ASYNC_LDS
            __builtin_amdgcn_global_load_async_to_lds_b128(
                GPTR(&wHiP[g]), LPTR(&lBhi[buf][lo]), 0, 0);
            __builtin_amdgcn_global_load_async_to_lds_b128(
                GPTR(&wLoP[g]), LPTR(&lBlo[buf][lo]), 0, 0);
#else
            *(v8bf*)&lBhi[buf][lo] = *(const v8bf*)&wHiP[g];
            *(v8bf*)&lBlo[buf][lo] = *(const v8bf*)&wLoP[g];
#endif
        }
    };

    auto storeA = [&](int buf, int k0) {       // normalize, split, -> LDS
        const int aColG = p * D_DIM + k0 + aCol;
        const float4 sc4 = *(const float4*)&scale[aColG];
        const float4 sh4 = *(const float4*)&shift[aColG];
        #pragma unroll
        for (int i = 0; i < 4; ++i) {
            const int r = aRow0 + 32 * i;
            const float n0 = aReg[i].x * sc4.x + sh4.x;
            const float n1 = aReg[i].y * sc4.y + sh4.y;
            const float n2 = aReg[i].z * sc4.z + sh4.z;
            const float n3 = aReg[i].w * sc4.w + sh4.w;
            v4bf hi = {(bf16_t)n0, (bf16_t)n1, (bf16_t)n2, (bf16_t)n3};
            v4bf lo = {(bf16_t)(n0 - (float)hi[0]), (bf16_t)(n1 - (float)hi[1]),
                       (bf16_t)(n2 - (float)hi[2]), (bf16_t)(n3 - (float)hi[3])};
            *(v4bf*)&lAhi[buf][r * LDT + aCol] = hi;
            *(v4bf*)&lAlo[buf][r * LDT + aCol] = lo;
        }
    };

    // ---- prologue: fill buffer 0 -----------------------------------------
    loadA(0);
    stageB(0, 0);
    storeA(0, 0);
#if USE_ASYNC_LDS
    asm volatile("s_wait_asynccnt 0x0" ::: "memory");
#endif
    __syncthreads();

    // ---- main pipeline ----------------------------------------------------
    for (int k = 0; k < NK; ++k) {
        const int cur = k & 1;
        const int nxt = cur ^ 1;
        const bool hasNext = (k + 1) < NK;

        // issue next tile's global traffic before the WMMA burst
        if (hasNext) {
            loadA((k + 1) * BK);
            stageB(nxt, (k + 1) * BK);
        }

        // ---- fragments: bf16 16x32 lane layout -> two 16B chunks + shuffle
        const bf16_t* As = lAhi[cur];
        const bf16_t* Al_ = lAlo[cur];
        const bf16_t* Bs = lBhi[cur];
        const bf16_t* Bl_ = lBlo[cur];

        v16bf Ah[4], Al[4];
        #pragma unroll
        for (int mf = 0; mf < 4; ++mf) {
            const int m = wm + mf * 16 + l16;
            v8bf h0 = *(const v8bf*)&As[m * LDT + kq];
            v8bf h1 = *(const v8bf*)&As[m * LDT + kq + 16];
            Ah[mf] = __builtin_shufflevector(h0, h1, 0,1,2,3,4,5,6,7,8,9,10,11,12,13,14,15);
            v8bf l0 = *(const v8bf*)&Al_[m * LDT + kq];
            v8bf l1 = *(const v8bf*)&Al_[m * LDT + kq + 16];
            Al[mf] = __builtin_shufflevector(l0, l1, 0,1,2,3,4,5,6,7,8,9,10,11,12,13,14,15);
        }
        #pragma unroll
        for (int nf = 0; nf < 2; ++nf) {
            const int e = wn + nf * 16 + l16;
            v8bf b0 = *(const v8bf*)&Bs[e * LDT + kq];
            v8bf b1 = *(const v8bf*)&Bs[e * LDT + kq + 16];
            v16bf Bh = __builtin_shufflevector(b0, b1, 0,1,2,3,4,5,6,7,8,9,10,11,12,13,14,15);
            v8bf c0 = *(const v8bf*)&Bl_[e * LDT + kq];
            v8bf c1 = *(const v8bf*)&Bl_[e * LDT + kq + 16];
            v16bf Bl = __builtin_shufflevector(c0, c1, 0,1,2,3,4,5,6,7,8,9,10,11,12,13,14,15);
            #pragma unroll
            for (int mf = 0; mf < 4; ++mf) {
                // split-fp32 product: hi*hi + hi*lo + lo*hi (f32 accumulate)
                acc[mf][nf] = __builtin_amdgcn_wmma_f32_16x16x32_bf16(
                    false, Ah[mf], false, Bh, (short)0, acc[mf][nf], false, false);
                acc[mf][nf] = __builtin_amdgcn_wmma_f32_16x16x32_bf16(
                    false, Ah[mf], false, Bl, (short)0, acc[mf][nf], false, false);
                acc[mf][nf] = __builtin_amdgcn_wmma_f32_16x16x32_bf16(
                    false, Al[mf], false, Bh, (short)0, acc[mf][nf], false, false);
            }
        }

        // finish writing the next buffer, then one barrier per K step
        if (hasNext)
            storeA(nxt, (k + 1) * BK);
#if USE_ASYNC_LDS
        asm volatile("s_wait_asynccnt 0x0" ::: "memory");
#endif
        __syncthreads();
    }

    // ---- epilogue: bias, sigmoid gate, residual. C layout: VGPR j holds
    // M=j (lanes 0-15) / M=j+8 (lanes 16-31), N = lane&15.
    const int mHalf = (lane >> 4) << 3;
    #pragma unroll
    for (int nf = 0; nf < 2; ++nf) {
        const int eP  = nInP + wn + nf * 16 + l16;   // index within partition
        const int col = p * D_DIM + eP;
        const float bs = bias[col];
        const float g  = gmm[eP];
        const float bt = bta[eP];
        #pragma unroll
        for (int mf = 0; mf < 4; ++mf) {
            #pragma unroll
            for (int j = 0; j < 8; ++j) {
                const int row = mBase + wm + mf * 16 + mHalf + j;
                const float h    = acc[mf][nf][j] + bs;
                const float sig  = 1.f / (1.f + __expf(-bt * h));
                const float gate = g + sig * (1.f - g);
                const size_t idx = (size_t)row * IN_F + col;
                out[idx] = gate * h + x[idx];
            }
        }
    }
}

// ---------------------------------------------------------------------------
// Launch. ws layout (floats unless noted):
//   psum   [B/256][IN]   psumsq [B/256][IN]   scale[IN]  shift[IN]
//   whiT   [P][D][D] bf16      wloT [P][D][D] bf16        (~10.1 MB total)
// ---------------------------------------------------------------------------
extern "C" void kernel_launch(void* const* d_in, const int* in_sizes, int n_in,
                              void* d_out, int out_size, void* d_ws, size_t ws_size,
                              hipStream_t stream)
{
    const float* x     = (const float*)d_in[0];
    const float* w     = (const float*)d_in[1];
    const float* bias  = (const float*)d_in[2];
    const float* bn_w  = (const float*)d_in[3];
    const float* bn_b  = (const float*)d_in[4];
    const float* gamma = (const float*)d_in[5];
    const float* beta  = (const float*)d_in[6];
    float* out = (float*)d_out;

    constexpr int CHUNKS = B_ROWS / 256;
    float* psum   = (float*)d_ws;
    float* psumsq = psum + (size_t)CHUNKS * IN_F;
    float* scale  = psumsq + (size_t)CHUNKS * IN_F;
    float* shift  = scale + IN_F;
    bf16_t* whiT  = (bf16_t*)(shift + IN_F);
    bf16_t* wloT  = whiT + (size_t)P_PART * D_DIM * D_DIM;

    bn_partial_kernel<<<dim3(IN_F / 256, CHUNKS), 256, 0, stream>>>(x, psum, psumsq);
    bn_finalize_kernel<<<IN_F / 256, 256, 0, stream>>>(psum, psumsq, bn_w, bn_b,
                                                       scale, shift);
    split_weights_kernel<<<(P_PART * D_DIM * D_DIM) / 256, 256, 0, stream>>>(
        w, whiT, wloT);
    fused_bn_blockgemm_gate_kernel<<<dim3(B_ROWS / 128, (P_PART * D_DIM) / 128),
                                     256, 0, stream>>>(
        x, whiT, wloT, scale, shift, bias, gamma, beta, out);
}
